// LSTM_28097676051118
// MI455X (gfx1250) — matrix operand order repaired and checked by reading, not compiled, not run on hardware
//
#include <hip/hip_runtime.h>
#include <math.h>

typedef __attribute__((ext_vector_type(16))) _Float16 v16h;
typedef __attribute__((ext_vector_type(8)))  _Float16 v8h;
typedef __attribute__((ext_vector_type(8)))  float    v8f;

#define B_ 256
#define T_ 512
#define D_ 256
#define H_ 1024
#define C_ 1000
#define NG 4096      // 4*H, gate-interleaved: n' = 4*j + gate (g,i,f,o)
#define KC 1280      // D + H
#define KB_GATES 40  // KC/32
#define NB_GATES 256 // NG/16
#define KB_PROJ  32  // H/32
#define NB_PROJ  64  // 1024/16 (N padded 1000->1024)

union V16 { v16h v; v8h h[2]; };

// ---------------------------------------------------------------------------
// Pack kernels
// ---------------------------------------------------------------------------
__global__ void lstm_pack_x(const float* __restrict__ x, _Float16* __restrict__ x16, int n) {
    int idx = blockIdx.x * blockDim.x + threadIdx.x;
    if (idx < n) x16[idx] = (_Float16)x[idx];
}

// Pack [W_x; W_h] (K=1280, N'=4096 gate-interleaved) into WMMA B-fragment order:
// tile = kb*NB + nb ; element v of lane l -> K = kb*32+(l>>4)*8+(v>>3)*16+(v&7), n' = nb*16+(l&15)
// n' = 4*j + gate  (gate: 0=g, 1=i, 2=f, 3=o)
__global__ void lstm_pack_wcat(const float* __restrict__ Wxg, const float* __restrict__ Whg,
                               const float* __restrict__ Wxi, const float* __restrict__ Whi,
                               const float* __restrict__ Wxf, const float* __restrict__ Whf,
                               const float* __restrict__ Wxo, const float* __restrict__ Who,
                               _Float16* __restrict__ Wp) {
    int idx = blockIdx.x * blockDim.x + threadIdx.x;
    if (idx >= KB_GATES * NB_GATES * 512) return;
    int tile = idx >> 9;
    int r    = idx & 511;
    int hh   = r >> 8;
    int rr   = r & 255;
    int l    = rr >> 3;
    int e    = rr & 7;
    int v    = hh * 8 + e;
    int kb   = tile / NB_GATES;
    int nb   = tile % NB_GATES;
    int K    = kb * 32 + ((l >> 4) * 8) + ((v >> 3) * 16) + (v & 7);
    int n    = nb * 16 + (l & 15);
    int j    = n >> 2;      // hidden unit
    int gate = n & 3;       // interleaved gate id
    const float* Wx = gate == 0 ? Wxg : gate == 1 ? Wxi : gate == 2 ? Wxf : Wxo;
    const float* Wh = gate == 0 ? Whg : gate == 1 ? Whi : gate == 2 ? Whf : Who;
    float w = (K < D_) ? Wx[(size_t)K * H_ + j] : Wh[(size_t)(K - D_) * H_ + j];
    Wp[idx] = (_Float16)w;
}

// Pack W_hp (K=1024, N=1000 padded to 1024), plain N order
__global__ void lstm_pack_whp(const float* __restrict__ Whp, _Float16* __restrict__ Wp) {
    int idx = blockIdx.x * blockDim.x + threadIdx.x;
    if (idx >= KB_PROJ * NB_PROJ * 512) return;
    int tile = idx >> 9;
    int r    = idx & 511;
    int hh   = r >> 8;
    int rr   = r & 255;
    int l    = rr >> 3;
    int e    = rr & 7;
    int v    = hh * 8 + e;
    int kb   = tile / NB_PROJ;
    int nb   = tile % NB_PROJ;
    int K    = kb * 32 + ((l >> 4) * 8) + ((v >> 3) * 16) + (v & 7);
    int n    = nb * 16 + (l & 15);
    float w  = (n < C_) ? Whp[(size_t)K * C_ + n] : 0.0f;
    Wp[idx] = (_Float16)w;
}

__global__ void lstm_init_state(float* __restrict__ c32, _Float16* __restrict__ h16, int n) {
    int idx = blockIdx.x * blockDim.x + threadIdx.x;
    if (idx < n) { c32[idx] = 0.0f; h16[idx] = (_Float16)0.0f; }
}

// ---------------------------------------------------------------------------
// Fused per-timestep kernel: GEMM ([x_t|h] @ Wcat, gate-interleaved N) + LSTM
// cell update. grid = (64 N'-blocks of 64, 4 M-blocks of 64); 256 thr = 8 waves.
// Wave (waveM 0..3, waveN 0..1) computes 16Mx32N via 2 WMMA accumulators.
// A tile is double-buffered in LDS and filled with async-to-LDS loads
// (ASYNCcnt) overlapped with the WMMAs of the current tile.
// ---------------------------------------------------------------------------
__global__ __launch_bounds__(256) void lstm_step_fused(
    const _Float16* __restrict__ x16,   // [B,T,D]
    const _Float16* __restrict__ hin,   // [B,H]
    _Float16* __restrict__ hout,        // [B,H]
    const _Float16* __restrict__ Wp,    // packed B tiles
    const float* __restrict__ bg, const float* __restrict__ bi,
    const float* __restrict__ bf, const float* __restrict__ bo,
    float* __restrict__ c32, int t) {
    // smem reused: during K-loop -> 2 x (64x32) f16 A tiles (8 KB)
    //              after K-loop -> 64 x 68 f32 gate-exchange (17408 B)
    __shared__ __align__(16) unsigned char smem[64 * 68 * 4];
    _Float16* As = (_Float16*)smem;
    float*    Gs = (float*)smem;

    const int tid   = threadIdx.x;
    const int lane  = tid & 31;
    const int wave  = tid >> 5;
    const int waveM = wave & 3;
    const int waveN = wave >> 2;
    const int bm    = blockIdx.y;   // 0..3
    const int bn    = blockIdx.x;   // 0..63

    // cooperative A-tile stager: thread -> (row 0..63, 8 contiguous cols)
    const int lrow = tid >> 2;
    const int lcol = (tid & 3) * 8;
    const int grow = bm * 64 + lrow;

    // per-lane A fragment read indices (ISA 16-bit A 16x32 layout)
    const int arow = waveM * 16 + (lane & 15);
    const int ako  = (lane >> 4) * 8;

    // async global->LDS stage of A tile kb into buffer buf
    auto stage = [&](int kb, int buf) {
        const _Float16* src = (kb < 8)
            ? x16 + (size_t)grow * (T_ * D_) + (size_t)t * D_ + kb * 32 + lcol
            : hin + (size_t)grow * H_ + (kb * 32 - D_) + lcol;
        unsigned ldsa = (unsigned)(size_t)&As[buf * 2048 + lrow * 32 + lcol];
        asm volatile("global_load_async_to_lds_b128 %0, %1, off"
                     :: "v"(ldsa), "v"((unsigned long long)(size_t)src)
                     : "memory");
    };

    v8f acc0 = {0,0,0,0,0,0,0,0};
    v8f acc1 = acc0;

    stage(0, 0);
    asm volatile("s_wait_asynccnt 0" ::: "memory");
    __syncthreads();

    for (int kb = 0; kb < KB_GATES; ++kb) {
        const int cur = kb & 1;
        V16 a;
        a.h[0] = *(const v8h*)&As[cur * 2048 + arow * 32 + ako];
        a.h[1] = *(const v8h*)&As[cur * 2048 + arow * 32 + ako + 16];

        if (kb + 1 < KB_GATES) stage(kb + 1, cur ^ 1);  // overlap with WMMAs

        const _Float16* wb = Wp + ((size_t)kb * NB_GATES + bn * 4 + waveN * 2) * 512;
        V16 b0, b1;
        b0.h[0] = *(const v8h*)(wb +   0 + lane * 8);
        b0.h[1] = *(const v8h*)(wb + 256 + lane * 8);
        b1.h[0] = *(const v8h*)(wb + 512 + lane * 8);
        b1.h[1] = *(const v8h*)(wb + 768 + lane * 8);

        acc0 = __builtin_amdgcn_wmma_f32_16x16x32_f16(false, a.v, false, b0.v, (short)0, acc0, false, false);
        acc1 = __builtin_amdgcn_wmma_f32_16x16x32_f16(false, a.v, false, b1.v, (short)0, acc1, false, false);

        asm volatile("s_wait_asynccnt 0" ::: "memory");
        __syncthreads();
    }

    // exchange gate pre-activations through LDS (cross-lane regroup)
    {
        const int rt = waveM * 16 + ((lane >> 4) * 8);
        const int ct = waveN * 32 + (lane & 15);
        #pragma unroll
        for (int r = 0; r < 8; ++r) {
            Gs[(rt + r) * 68 + ct]      = acc0[r];
            Gs[(rt + r) * 68 + ct + 16] = acc1[r];
        }
    }
    __syncthreads();

    // fused LSTM cell update: this WG owns rows [bm*64,+64) x hidden [bn*16,+16)
    #pragma unroll
    for (int kc = 0; kc < 4; ++kc) {
        int cell = tid + kc * 256;      // 0..1023
        int rt   = cell >> 4;           // 0..63
        int jj   = cell & 15;           // 0..15
        int j    = bn * 16 + jj;        // global hidden index
        int row  = bm * 64 + rt;        // global batch row
        float g = Gs[rt * 68 + jj * 4 + 0] + bg[j];
        float i = Gs[rt * 68 + jj * 4 + 1] + bi[j];
        float f = Gs[rt * 68 + jj * 4 + 2] + bf[j];
        float o = Gs[rt * 68 + jj * 4 + 3] + bo[j];
        g = tanhf(g);
        i = 1.0f / (1.0f + __expf(-i));
        f = 1.0f / (1.0f + __expf(-f));
        o = 1.0f / (1.0f + __expf(-o));
        size_t ci = (size_t)row * H_ + j;
        float cn = g * i + c32[ci] * f;
        c32[ci]  = cn;
        hout[ci] = (_Float16)(tanhf(cn) * o);
    }
}

// ---------------------------------------------------------------------------
// Final projection: out[256,1000] = h @ W_hp + b_p (N padded to 1024)
// grid = (16 N-blocks of 64, 4 M-blocks of 64); same async double-buffering
// ---------------------------------------------------------------------------
__global__ __launch_bounds__(256) void lstm_proj(
    const _Float16* __restrict__ h16,
    const _Float16* __restrict__ Wp,
    const float* __restrict__ bp,
    float* __restrict__ out) {
    __shared__ __align__(16) _Float16 As[2 * 64 * 32];

    const int tid   = threadIdx.x;
    const int lane  = tid & 31;
    const int wave  = tid >> 5;
    const int waveM = wave & 3;
    const int waveN = wave >> 2;
    const int bm    = blockIdx.y;
    const int bn    = blockIdx.x;

    const int lrow = tid >> 2;
    const int lcol = (tid & 3) * 8;
    const int grow = bm * 64 + lrow;

    const int arow = waveM * 16 + (lane & 15);
    const int ako  = (lane >> 4) * 8;

    auto stage = [&](int kb, int buf) {
        const _Float16* src = h16 + (size_t)grow * H_ + kb * 32 + lcol;
        unsigned ldsa = (unsigned)(size_t)&As[buf * 2048 + lrow * 32 + lcol];
        asm volatile("global_load_async_to_lds_b128 %0, %1, off"
                     :: "v"(ldsa), "v"((unsigned long long)(size_t)src)
                     : "memory");
    };

    v8f acc0 = {0,0,0,0,0,0,0,0};
    v8f acc1 = acc0;

    stage(0, 0);
    asm volatile("s_wait_asynccnt 0" ::: "memory");
    __syncthreads();

    for (int kb = 0; kb < KB_PROJ; ++kb) {
        const int cur = kb & 1;
        V16 a;
        a.h[0] = *(const v8h*)&As[cur * 2048 + arow * 32 + ako];
        a.h[1] = *(const v8h*)&As[cur * 2048 + arow * 32 + ako + 16];

        if (kb + 1 < KB_PROJ) stage(kb + 1, cur ^ 1);

        const _Float16* wb = Wp + ((size_t)kb * NB_PROJ + bn * 4 + waveN * 2) * 512;
        V16 b0, b1;
        b0.h[0] = *(const v8h*)(wb +   0 + lane * 8);
        b0.h[1] = *(const v8h*)(wb + 256 + lane * 8);
        b1.h[0] = *(const v8h*)(wb + 512 + lane * 8);
        b1.h[1] = *(const v8h*)(wb + 768 + lane * 8);

        acc0 = __builtin_amdgcn_wmma_f32_16x16x32_f16(false, a.v, false, b0.v, (short)0, acc0, false, false);
        acc1 = __builtin_amdgcn_wmma_f32_16x16x32_f16(false, a.v, false, b1.v, (short)0, acc1, false, false);

        asm volatile("s_wait_asynccnt 0" ::: "memory");
        __syncthreads();
    }

    const int srow = bm * 64 + waveM * 16 + ((lane >> 4) * 8);
    const int scol = bn * 64 + waveN * 32 + (lane & 15);
    #pragma unroll
    for (int r = 0; r < 8; ++r) {
        int row = srow + r;
        if (scol      < C_) out[(size_t)row * C_ + scol]      = acc0[r] + bp[scol];
        if (scol + 16 < C_) out[(size_t)row * C_ + scol + 16] = acc1[r] + bp[scol + 16];
    }
}

// ---------------------------------------------------------------------------
// Host launcher
// ---------------------------------------------------------------------------
extern "C" void kernel_launch(void* const* d_in, const int* in_sizes, int n_in,
                              void* d_out, int out_size, void* d_ws, size_t ws_size,
                              hipStream_t stream) {
    const float* x   = (const float*)d_in[0];
    const float* Wxg = (const float*)d_in[1];
    const float* Whg = (const float*)d_in[2];
    const float* bg  = (const float*)d_in[3];
    const float* Wxi = (const float*)d_in[4];
    const float* Whi = (const float*)d_in[5];
    const float* bi  = (const float*)d_in[6];
    const float* Wxf = (const float*)d_in[7];
    const float* Whf = (const float*)d_in[8];
    const float* bf  = (const float*)d_in[9];
    const float* Wxo = (const float*)d_in[10];
    const float* Who = (const float*)d_in[11];
    const float* bo  = (const float*)d_in[12];
    const float* Whp = (const float*)d_in[13];
    const float* bp  = (const float*)d_in[14];

    char* ws = (char*)d_ws;
    // workspace layout (all 16B aligned), total ~78 MB
    _Float16* x16    = (_Float16*)(ws);                       // 67,108,864 B
    _Float16* wcat16 = (_Float16*)(ws + 67108864);            // 10,485,760 B
    _Float16* whp16  = (_Float16*)(ws + 77594624);            //  2,097,152 B
    float*    c32    = (float*)   (ws + 79691776);            //  1,048,576 B
    _Float16* h0     = (_Float16*)(ws + 80740352);            //    524,288 B
    _Float16* h1     = (_Float16*)(ws + 81264640);            //    524,288 B

    const int NX = B_ * T_ * D_;
    lstm_pack_x<<<(NX + 255) / 256, 256, 0, stream>>>(x, x16, NX);
    lstm_pack_wcat<<<(KB_GATES * NB_GATES * 512 + 255) / 256, 256, 0, stream>>>(
        Wxg, Whg, Wxi, Whi, Wxf, Whf, Wxo, Who, wcat16);
    lstm_pack_whp<<<(KB_PROJ * NB_PROJ * 512 + 255) / 256, 256, 0, stream>>>(Whp, whp16);
    lstm_init_state<<<(B_ * H_ + 255) / 256, 256, 0, stream>>>(c32, h0, B_ * H_);

    _Float16* hbuf[2] = { h0, h1 };
    for (int t = 0; t < T_; ++t) {
        lstm_step_fused<<<dim3(64, 4), 256, 0, stream>>>(
            x16, hbuf[t & 1], hbuf[(t + 1) & 1], wcat16,
            bg, bi, bf, bo, c32, t);
    }
    // T_=512 is even -> final h lands back in h0
    lstm_proj<<<dim3(16, 4), 256, 0, stream>>>(h0, whp16, bp, (float*)d_out);
}
